// WindowAttention_34909494181960
// MI455X (gfx1250) — matrix operand order repaired and checked by reading, not compiled, not run on hardware
//
#include <hip/hip_runtime.h>

#define C_DIM 256
#define N_TOK 16384
#define B_SZ  8
#define H_NUM 4
#define KSLICES 16

typedef __attribute__((ext_vector_type(16))) __bf16        v16bf;
typedef __attribute__((ext_vector_type(8)))  float         v8f;
typedef __attribute__((ext_vector_type(8)))  unsigned short us8;
typedef __attribute__((ext_vector_type(4)))  unsigned short us4;
typedef __attribute__((ext_vector_type(4)))  float         f4;

union BfPack { struct { us8 lo, hi; } h; v16bf v; };

__device__ __forceinline__ unsigned short f2bf(float f) {
  unsigned int u = __builtin_bit_cast(unsigned int, f);
  u += 0x7fffu + ((u >> 16) & 1u);          // round-to-nearest-even
  return (unsigned short)(u >> 16);
}
__device__ __forceinline__ float bf2f(unsigned short s) {
  return __builtin_bit_cast(float, (unsigned int)s << 16);
}
__device__ __forceinline__ v8f wmma_bf16(v16bf a, v16bf b, v8f c) {
  return __builtin_amdgcn_wmma_f32_16x16x32_bf16(false, a, false, b, (short)0, c,
                                                 false, false);
}

// ---------------------------------------------------------------------------
// Kernel 0: convert kv_w (512x256) and proj_w (256x256) f32 -> bf16 in ws
// ---------------------------------------------------------------------------
__global__ void k_convert(const float* __restrict__ kvw, const float* __restrict__ pjw,
                          unsigned short* __restrict__ kvw_bf,
                          unsigned short* __restrict__ pjw_bf) {
  int i = blockIdx.x * 256 + threadIdx.x;
  if (i < 512 * 256) kvw_bf[i] = f2bf(kvw[i]);
  if (i < 256 * 256) pjw_bf[i] = f2bf(pjw[i]);
}

// ---------------------------------------------------------------------------
// Kernel 1: kv projection. y[m][j] = x[m][:] . kv_w[j][:] + kv_b[j]
// j = two*256 + h*64 + d.  two==0 -> k_ws[(bh*64+d)*N + n]   (bf16)
//                          two==1 -> v_t [(bh*N + n)*64 + d] (bf16)
// Block: 32 M-rows x 256 cols; 8 waves, each wave = (m-tile, 4 col-tiles)
// => 4 WMMAs per barrier pair. A staged in LDS in WMMA lane order; fill is
// one float4 load + one b64 ds store per thread per K-step.
// ---------------------------------------------------------------------------
__global__ void k_kvproj(const float* __restrict__ x,
                         const unsigned short* __restrict__ kvw_bf,
                         const float* __restrict__ kv_b,
                         unsigned short* __restrict__ k_ws,
                         unsigned short* __restrict__ v_t) {
  __shared__ __align__(16) unsigned short A_lds[1024];  // 2 m-tiles x (16x32) bf16
  const int t = threadIdx.x;
  const int lane = t & 31, w = t >> 5;
  const int r = lane & 15, hi = lane >> 4;
  const int m0 = blockIdx.x * 32;
  const int mt = w >> 2;                 // wave's m-tile (0/1)
  const int cg = w & 3;                  // wave's column group (4 tiles of 16)
  const int j0 = blockIdx.y * 256 + cg * 64;

  // fill decomposition: thread -> (row 0..31, k-quad 0..7), lane-order dest
  const int frow = t >> 3;
  const int fkq  = t & 7;
  const int flane = (frow & 15) + ((fkq >> 1) & 1) * 16;
  const int fj    = (fkq & 1) * 4 + ((fkq >> 2) & 1) * 8;
  const int fdst  = (frow >> 4) * 512 + flane * 16 + fj;
  const float* fsrc = x + (size_t)(m0 + frow) * C_DIM + fkq * 4;

  v8f acc[4];
#pragma unroll
  for (int i = 0; i < 4; ++i) acc[i] = v8f{0.f, 0.f, 0.f, 0.f, 0.f, 0.f, 0.f, 0.f};

  for (int k0 = 0; k0 < 256; k0 += 32) {
    __syncthreads();
    f4 xv = *(const f4*)(fsrc + k0);
    us4 pk = {f2bf(xv[0]), f2bf(xv[1]), f2bf(xv[2]), f2bf(xv[3])};
    *(us4*)&A_lds[fdst] = pk;
    __syncthreads();

    BfPack a;
    a.h.lo = *(const us8*)&A_lds[mt * 512 + lane * 16];
    a.h.hi = *(const us8*)&A_lds[mt * 512 + lane * 16 + 8];
#pragma unroll
    for (int ti = 0; ti < 4; ++ti) {
      BfPack b;
      const unsigned short* bp = kvw_bf + (size_t)(j0 + ti * 16 + r) * 256 + k0 + hi * 8;
      b.h.lo = *(const us8*)bp;
      b.h.hi = *(const us8*)(bp + 16);
      acc[ti] = wmma_bf16(a.v, b.v, acc[ti]);
    }
  }

  const int mrow0 = m0 + mt * 16;
  const int b = mrow0 >> 14;
  const int nbase = (mrow0 & (N_TOK - 1)) + hi * 8;
#pragma unroll
  for (int ti = 0; ti < 4; ++ti) {
    const int col = j0 + ti * 16 + r;
    const float bias = kv_b[col];
    const int two = col >> 8, h = (col >> 6) & 3, dd = col & 63;
    const int bh = b * H_NUM + h;
    v8f c = acc[ti];
    if (two == 0) {
      unsigned short* dst = k_ws + (size_t)(bh * 64 + dd) * N_TOK + nbase;
      us4 s0 = {f2bf(c[0] + bias), f2bf(c[1] + bias), f2bf(c[2] + bias), f2bf(c[3] + bias)};
      us4 s1 = {f2bf(c[4] + bias), f2bf(c[5] + bias), f2bf(c[6] + bias), f2bf(c[7] + bias)};
      *(us4*)dst = s0;
      *(us4*)(dst + 4) = s1;
    } else {
#pragma unroll
      for (int i = 0; i < 8; ++i)
        v_t[((size_t)bh * N_TOK + nbase + i) * 64 + dd] = f2bf(c[i] + bias);
    }
  }
}

// ---------------------------------------------------------------------------
// Kernel 2: inv_norm[row] = 1/max(||k_ws[row][:]||, eps), row = bh*64+d (2048)
// ---------------------------------------------------------------------------
__global__ void k_rownorm(const unsigned short* __restrict__ k_ws,
                          float* __restrict__ invn) {
  __shared__ float red[256];
  const int row = blockIdx.x;
  const unsigned short* p = k_ws + (size_t)row * N_TOK;
  float s = 0.f;
  for (int i = threadIdx.x; i < N_TOK; i += 256) {
    float v = bf2f(p[i]);
    s += v * v;
  }
  red[threadIdx.x] = s;
  __syncthreads();
  for (int st = 128; st > 0; st >>= 1) {
    if (threadIdx.x < st) red[threadIdx.x] += red[threadIdx.x + st];
    __syncthreads();
  }
  if (threadIdx.x == 0) invn[row] = 1.f / fmaxf(sqrtf(red[0]), 1e-12f);
}

// ---------------------------------------------------------------------------
// Kernel 3: transpose const (B,N,C) -> q_ws[(b*256+c)*N + n] bf16 (LDS 64x64)
// ---------------------------------------------------------------------------
__global__ void k_qtrans(const float* __restrict__ cst,
                         unsigned short* __restrict__ q_ws) {
  __shared__ float tile[64][65];
  const int b = blockIdx.z, n0 = blockIdx.x * 64, c0 = blockIdx.y * 64;
  const int t = threadIdx.x, cc = t & 63, r0 = t >> 6;
#pragma unroll
  for (int j = 0; j < 16; ++j) {
    int row = r0 + j * 4;
    tile[row][cc] = cst[((size_t)b * N_TOK + n0 + row) * C_DIM + c0 + cc];
  }
  __syncthreads();
#pragma unroll
  for (int j = 0; j < 16; ++j) {
    int crow = r0 + j * 4;
    q_ws[((size_t)b * 256 + c0 + crow) * N_TOK + n0 + cc] = f2bf(tile[cc][crow]);
  }
}

// ---------------------------------------------------------------------------
// Kernel 4a: qk partials. Grid (bh=32, slice=16): each block reduces a K-slice
// of 1024 over n. Wave w owns (d-tile = w>>1, e-half = w&1). Partials go to
// pqk[(bh*16+slice)*4096 + d*64+e] so the reduction can be done in a FIXED
// order (deterministic, no float atomics) by kernel 4b.
// ---------------------------------------------------------------------------
__global__ void k_qk_partial(const unsigned short* __restrict__ q_ws,
                             const unsigned short* __restrict__ k_ws,
                             float* __restrict__ pqk) {
  const int bh = blockIdx.x;
  const int sl = blockIdx.y;
  const int t = threadIdx.x, lane = t & 31, w = t >> 5;
  const int r = lane & 15, hi = lane >> 4;
  const int dt = w >> 1;        // d-tile 0..3
  const int eh = w & 1;         // e-half 0..1
  const unsigned short* qbase = q_ws + (size_t)bh * 64 * N_TOK + (size_t)(dt * 16 + r) * N_TOK;
  const unsigned short* kbase = k_ws + (size_t)bh * 64 * N_TOK;

  v8f acc0 = {0.f, 0.f, 0.f, 0.f, 0.f, 0.f, 0.f, 0.f};
  v8f acc1 = acc0;
  const int nstart = sl * (N_TOK / KSLICES);
  const int nend = nstart + (N_TOK / KSLICES);
  for (int n0 = nstart; n0 < nend; n0 += 32) {
    BfPack a, b0, b1;
    const unsigned short* ap = qbase + n0 + hi * 8;
    a.h.lo = *(const us8*)ap;
    a.h.hi = *(const us8*)(ap + 16);
    const unsigned short* bp0 = kbase + (size_t)(eh * 32 + r) * N_TOK + n0 + hi * 8;
    const unsigned short* bp1 = kbase + (size_t)(eh * 32 + 16 + r) * N_TOK + n0 + hi * 8;
    b0.h.lo = *(const us8*)bp0;  b0.h.hi = *(const us8*)(bp0 + 16);
    b1.h.lo = *(const us8*)bp1;  b1.h.hi = *(const us8*)(bp1 + 16);
    __builtin_prefetch(ap + 32, 0, 3);     // global_prefetch_b8: next K tile
    __builtin_prefetch(bp0 + 32, 0, 3);
    acc0 = wmma_bf16(a.v, b0.v, acc0);
    acc1 = wmma_bf16(a.v, b1.v, acc1);
  }
  float* dst = pqk + ((size_t)bh * KSLICES + sl) * 4096;
#pragma unroll
  for (int i = 0; i < 8; ++i) {
    int d = dt * 16 + hi * 8 + i;
    dst[d * 64 + eh * 32 + r]      = acc0[i];
    dst[d * 64 + eh * 32 + 16 + r] = acc1[i];
  }
}

// ---------------------------------------------------------------------------
// Kernel 4b: fixed-order slice reduction + inv-norm scaling + fused dual
// softmax: attn = softmax(0.5*qk/8 + 0.5*softmax(qk)). bf16 out for GEMM3.
// ---------------------------------------------------------------------------
__global__ void k_softmax(const float* __restrict__ pqk,
                          const float* __restrict__ invn,
                          unsigned short* __restrict__ attn_ws) {
  __shared__ float qk[64 * 64];
  const int bh = blockIdx.x;
  const int t = threadIdx.x;
  const float* base = pqk + (size_t)bh * KSLICES * 4096;
  for (int idx = t; idx < 4096; idx += 256) {
    float s = 0.f;
#pragma unroll
    for (int k = 0; k < KSLICES; ++k) s += base[k * 4096 + idx];
    qk[idx] = s;
  }
  __syncthreads();

  if (t < 64) {                          // row d = t
    float v[64];
    const float* ip = invn + bh * 64;
#pragma unroll
    for (int e = 0; e < 64; ++e) v[e] = qk[t * 64 + e] * ip[e];
    float m1 = -1e30f;
#pragma unroll
    for (int e = 0; e < 64; ++e) m1 = fmaxf(m1, v[e]);
    float s1 = 0.f;
#pragma unroll
    for (int e = 0; e < 64; ++e) s1 += __expf(v[e] - m1);
    float r1 = 1.f / s1;
#pragma unroll
    for (int e = 0; e < 64; ++e)         // 0.5*qk/8 + 0.5*softmax(qk)
      v[e] = v[e] * 0.0625f + 0.5f * (__expf(v[e] - m1) * r1);
    float m2 = -1e30f;
#pragma unroll
    for (int e = 0; e < 64; ++e) m2 = fmaxf(m2, v[e]);
    float s2 = 0.f;
#pragma unroll
    for (int e = 0; e < 64; ++e) s2 += __expf(v[e] - m2);
    float r2 = 1.f / s2;
#pragma unroll
    for (int e = 0; e < 64; ++e)
      attn_ws[bh * 4096 + t * 64 + e] = f2bf(__expf(v[e] - m2) * r2);
  }
}

// ---------------------------------------------------------------------------
// Kernel 5: out_T[n][d] = sum_e v_t[n][e] * attn[d][e]  (transposed GEMM), then
// scatter through the torch-faithful reshape into out2 (row-major (b',n',c')).
// Block: (bh, 256-n chunk); wave w: n-rows [w*32,w*32+32), all 4 d-tiles.
// ---------------------------------------------------------------------------
__global__ void k_av(const unsigned short* __restrict__ v_t,
                     const unsigned short* __restrict__ attn_ws,
                     unsigned short* __restrict__ out2) {
  const int bh = blockIdx.x;
  const int t = threadIdx.x, lane = t & 31, w = t >> 5;
  const int r = lane & 15, hi = lane >> 4;
  const int nbase = blockIdx.y * 256 + w * 32;
  const unsigned short* vb = v_t + (size_t)bh * N_TOK * 64;
  const unsigned short* ab = attn_ws + bh * 4096;

  v8f acc[8];
#pragma unroll
  for (int i = 0; i < 8; ++i) acc[i] = v8f{0.f, 0.f, 0.f, 0.f, 0.f, 0.f, 0.f, 0.f};

#pragma unroll
  for (int k0 = 0; k0 < 64; k0 += 32) {
    BfPack A[2], Bm[4];
#pragma unroll
    for (int mt = 0; mt < 2; ++mt) {
      const unsigned short* p = vb + (size_t)(nbase + mt * 16 + r) * 64 + k0 + hi * 8;
      A[mt].h.lo = *(const us8*)p;
      A[mt].h.hi = *(const us8*)(p + 16);
    }
#pragma unroll
    for (int dtile = 0; dtile < 4; ++dtile) {
      const unsigned short* p = ab + (dtile * 16 + r) * 64 + k0 + hi * 8;
      Bm[dtile].h.lo = *(const us8*)p;
      Bm[dtile].h.hi = *(const us8*)(p + 16);
    }
#pragma unroll
    for (int mt = 0; mt < 2; ++mt)
#pragma unroll
      for (int dtile = 0; dtile < 4; ++dtile)
        acc[mt * 4 + dtile] = wmma_bf16(A[mt].v, Bm[dtile].v, acc[mt * 4 + dtile]);
  }

  const int b = bh >> 2, h = bh & 3;
  const int bp = 2 * h + (b >> 2);                     // b' of buggy reshape
#pragma unroll
  for (int mt = 0; mt < 2; ++mt) {
    const int n0 = nbase + mt * 16;
#pragma unroll
    for (int dtile = 0; dtile < 4; ++dtile) {
      const int d = dtile * 16 + r;
      size_t base = (size_t)bp * (N_TOK * C_DIM) +
                    (size_t)((b & 3) * 4096 + d * 64 + (n0 >> 8)) * 256 +
                    (n0 & 255) + hi * 8;
      v8f c = acc[mt * 4 + dtile];
      us4 s0 = {f2bf(c[0]), f2bf(c[1]), f2bf(c[2]), f2bf(c[3])};
      us4 s1 = {f2bf(c[4]), f2bf(c[5]), f2bf(c[6]), f2bf(c[7])};
      *(us4*)(out2 + base) = s0;
      *(us4*)(out2 + base + 4) = s1;
    }
  }
}

// ---------------------------------------------------------------------------
// Kernel 6: final projection (transposed): res_T[cout][m] = proj_w . out2_row.
// Each wave owns TWO m-tiles sharing one A (proj_w) operand per K-step:
// 2 WMMAs per A load pair. Stores are contiguous f32 float4 per lane. +proj_b.
// ---------------------------------------------------------------------------
__global__ void k_proj(const unsigned short* __restrict__ out2,
                       const unsigned short* __restrict__ pjw_bf,
                       const float* __restrict__ pjb,
                       float* __restrict__ out) {
  const int t = threadIdx.x, lane = t & 31, w = t >> 5;
  const int r = lane & 15, hi = lane >> 4;
  const int m0 = blockIdx.x * 256 + w * 32;   // two m-tiles: m0, m0+16
  const int c0 = blockIdx.y * 16;
  v8f acc0 = {0.f, 0.f, 0.f, 0.f, 0.f, 0.f, 0.f, 0.f};
  v8f acc1 = acc0;

  for (int k0 = 0; k0 < 256; k0 += 32) {
    BfPack a, b0, b1;
    const unsigned short* ap = pjw_bf + (size_t)(c0 + r) * 256 + k0 + hi * 8;
    a.h.lo = *(const us8*)ap;
    a.h.hi = *(const us8*)(ap + 16);
    const unsigned short* bp0 = out2 + (size_t)(m0 + r) * 256 + k0 + hi * 8;
    const unsigned short* bp1 = out2 + (size_t)(m0 + 16 + r) * 256 + k0 + hi * 8;
    b0.h.lo = *(const us8*)bp0;  b0.h.hi = *(const us8*)(bp0 + 16);
    b1.h.lo = *(const us8*)bp1;  b1.h.hi = *(const us8*)(bp1 + 16);
    acc0 = wmma_bf16(a.v, b0.v, acc0);
    acc1 = wmma_bf16(a.v, b1.v, acc1);
  }

  const int cb = c0 + hi * 8;
  const float p0 = pjb[cb + 0], p1 = pjb[cb + 1], p2 = pjb[cb + 2], p3 = pjb[cb + 3];
  const float p4 = pjb[cb + 4], p5 = pjb[cb + 5], p6 = pjb[cb + 6], p7 = pjb[cb + 7];
  {
    f4 s0 = {acc0[0] + p0, acc0[1] + p1, acc0[2] + p2, acc0[3] + p3};
    f4 s1 = {acc0[4] + p4, acc0[5] + p5, acc0[6] + p6, acc0[7] + p7};
    float* dst = out + (size_t)(m0 + r) * 256 + cb;
    *(f4*)dst = s0;
    *(f4*)(dst + 4) = s1;
  }
  {
    f4 s0 = {acc1[0] + p0, acc1[1] + p1, acc1[2] + p2, acc1[3] + p3};
    f4 s1 = {acc1[4] + p4, acc1[5] + p5, acc1[6] + p6, acc1[7] + p7};
    float* dst = out + (size_t)(m0 + 16 + r) * 256 + cb;
    *(f4*)dst = s0;
    *(f4*)(dst + 4) = s1;
  }
}

// ---------------------------------------------------------------------------
extern "C" void kernel_launch(void* const* d_in, const int* in_sizes, int n_in,
                              void* d_out, int out_size, void* d_ws, size_t ws_size,
                              hipStream_t stream) {
  const float* x   = (const float*)d_in[0];
  const float* cst = (const float*)d_in[1];
  const float* kvw = (const float*)d_in[2];
  const float* kvb = (const float*)d_in[3];
  const float* pjw = (const float*)d_in[4];
  const float* pjb = (const float*)d_in[5];
  float* out = (float*)d_out;

  char* ws = (char*)d_ws;
  size_t off = 0;
  auto carve = [&](size_t bytes) {
    char* p = ws + off;
    off += (bytes + 255) & ~(size_t)255;
    return p;
  };
  unsigned short* kvw_bf = (unsigned short*)carve((size_t)512 * 256 * 2);
  unsigned short* pjw_bf = (unsigned short*)carve((size_t)256 * 256 * 2);
  unsigned short* k_ws   = (unsigned short*)carve((size_t)2048 * N_TOK * 2); // [bh*64+d][n]
  unsigned short* v_t    = (unsigned short*)carve((size_t)2048 * N_TOK * 2); // [bh][n][e]
  unsigned short* q_ws   = (unsigned short*)carve((size_t)2048 * N_TOK * 2); // [bh*64+d][n]
  float*          invn   = (float*)carve((size_t)2048 * 4);
  unsigned short* attn   = (unsigned short*)carve((size_t)32 * 4096 * 2);
  float*          pqk    = (float*)carve((size_t)32 * KSLICES * 4096 * 4);
  unsigned short* out2   = q_ws;   // q consumed by k_qk_partial before k_av writes

  k_convert   <<<dim3(512),         dim3(256), 0, stream>>>(kvw, pjw, kvw_bf, pjw_bf);
  k_kvproj    <<<dim3(4096, 2),     dim3(256), 0, stream>>>(x, kvw_bf, kvb, k_ws, v_t);
  k_rownorm   <<<dim3(2048),        dim3(256), 0, stream>>>(k_ws, invn);
  k_qtrans    <<<dim3(256, 4, 8),   dim3(256), 0, stream>>>(cst, q_ws);
  k_qk_partial<<<dim3(32, KSLICES), dim3(256), 0, stream>>>(q_ws, k_ws, pqk);
  k_softmax   <<<dim3(32),          dim3(256), 0, stream>>>(pqk, invn, attn);
  k_av        <<<dim3(32, 64),      dim3(256), 0, stream>>>(v_t, attn, out2);
  k_proj      <<<dim3(512, 16),     dim3(256), 0, stream>>>(out2, pjw_bf, pjb, out);
}
